// CC_module_H_66520453481211
// MI455X (gfx1250) — matrix operand order repaired and checked by reading, not compiled, not run on hardware
//
#include <hip/hip_runtime.h>

// ---------------------------------------------------------------------------
// MI455X (gfx1250) fused criss-cross H-attention.
//
// One workgroup per (b, w) column. Everything per-column lives in LDS
// (241 KB of the WGP's 320 KB -- a CDNA5-only fusion; CDNA4's 64 KB LDS
// cannot hold V[512x96]bf16 + X + Q/K + E + A simultaneously).
//
// Phases (16 waves, 512 threads, wave32; 4 waves/SIMD for latency hiding):
//   0: stage x[b,:,:,w] -> LDS XT[h][c] as bf16          (HBM read #1 of x)
//   1: QKV projection GEMMs via v_wmma_f32_16x16x32_bf16
//      Q,K stored transposed QT/KT[h][c8]; V stored VS[c][h]
//   2: E = Q.K^T  (96x96, K=64) -> LDS f32 (overlays dead XT)
//   3: row softmax (wave32 butterfly shuffles) -> A bf16
//   4: Out = V.A^T (512x96, K=96) via WMMA; out = gamma*acc + x (x read #2)
//
// Roofline: 56.6 GFLOP, ~453 MB mandatory HBM (~19 us @ 23.3 TB/s); bf16
// WMMA with f32 accumulation sits at the compute/BW knee, so the matrix
// path is entirely WMMA.  All fragment loads are batched ahead of each
// wmma chain so the compiler can issue one ds clause and cover LDS latency
// with preceding wmmas (partial s_wait_dscnt) instead of dscnt==0 stalls.
// ---------------------------------------------------------------------------

typedef __attribute__((ext_vector_type(16))) __bf16 v16bf;
typedef __attribute__((ext_vector_type(8)))  __bf16 v8bf;
typedef __attribute__((ext_vector_type(8)))  float  v8f;

#define NB   8
#define NC   512
#define NC8  64
#define NH   96
#define NW   96
#define NTHREADS 512
#define NWAVES   16

// ---- LDS layout (bytes). Row pitches padded to 32B multiples (wmma frag
// loads are 16B/32B vectors) and skewed vs the 256B bank period. ----
#define XP 528   // XT: [96 h][528] bf16   (cols 0..511 = c)
#define QP 80    // QT/KT: [96 h][80] bf16 (cols 0..63 = c8)
#define VP 112   // VS: [512 c][112] bf16  (cols 0..95 = h/j)
#define EP 100   // ES: [96 i][100] f32    (cols 0..95 = j)
#define AP 112   // AT: [96 i][112] bf16   (cols 0..95 = j)

#define XT_OFF 0
#define ES_OFF 0                       // overlays XT (dead after phase 1)
#define AT_OFF 38400                   // ES is 96*100*4 = 38400 B
#define QT_OFF 101376                  // XT is 96*528*2 = 101376 B
#define KT_OFF (QT_OFF + 96*QP*2)      // 116736
#define VS_OFF (KT_OFF + 96*QP*2)      // 132096
#define SMEM_BYTES (VS_OFF + 512*VP*2) // 246784 B  (<= 320 KB WGP LDS)

// A-fragment (16x32 bf16) from an LDS row, contiguous k order (B frags use the same):
// lane L holds row M=L%16 with K = kh*16+0..15, kh = lane>>4.
__device__ inline v16bf load_afrag_lds(const __bf16* row, int kk, int khalf) {
  v8bf lo = *(const v8bf*)(row + kk + khalf * 16);
  v8bf hi = *(const v8bf*)(row + kk + khalf * 16 + 8);
  return __builtin_shufflevector(lo, hi, 0,1,2,3,4,5,6,7,8,9,10,11,12,13,14,15);
}

// A-fragment from a global f32 weight row (convert f32 -> bf16 in flight).
__device__ inline v16bf load_afrag_w(const float* wrow, int kk, int khalf) {
  v16bf a;
#pragma unroll
  for (int e = 0; e < 8; ++e) a[e]     = (__bf16)wrow[kk + khalf * 16 + e];
#pragma unroll
  for (int e = 0; e < 8; ++e) a[8 + e] = (__bf16)wrow[kk + khalf * 16 + 8 + e];
  return a;
}

__global__ __launch_bounds__(NTHREADS, 1)
void cca_h_fused(const float* __restrict__ x,
                 const float* __restrict__ Wq, const float* __restrict__ bq,
                 const float* __restrict__ Wk, const float* __restrict__ bk,
                 const float* __restrict__ Wv, const float* __restrict__ bv,
                 const float* __restrict__ gamma,
                 float* __restrict__ out)
{
  extern __shared__ __align__(32) unsigned char smem[];
  __bf16* XT = (__bf16*)(smem + XT_OFF);
  __bf16* QT = (__bf16*)(smem + QT_OFF);
  __bf16* KT = (__bf16*)(smem + KT_OFF);
  __bf16* VS = (__bf16*)(smem + VS_OFF);
  float*  ES = (float*) (smem + ES_OFF);
  __bf16* AT = (__bf16*)(smem + AT_OFF);

  const int w    = blockIdx.x;
  const int b    = blockIdx.y;
  const int tid  = threadIdx.x;
  const int wave = tid >> 5;         // wave32
  const int lane = tid & 31;
  const int l16  = lane & 15;
  const int khf  = lane >> 4;
  const float g  = gamma[0];

  // ---------------- phase 0: stage x column -> XT[h][c] (bf16) ------------
  {
    const float* xb = x + (long)b * NC * NH * NW + w;
    for (int idx = tid; idx < NC * NH; idx += NTHREADS) {
      int h = idx % NH;              // h-fast: 384B-stride loads, L2-friendly
      int c = idx / NH;              // across neighboring-w workgroups
      XT[h * XP + c] = (__bf16)xb[(long)c * (NH * NW) + h * NW];
    }
  }
  __syncthreads();

  // ---------------- phase 1: QKV projections (WMMA bf16) ------------------
  // 640 output rows (64 q | 64 k | 512 v) x 96 cols; K = 512.
  for (int rt = wave; rt < 40; rt += NWAVES) {
    int o0 = rt * 16;
    const float *Wm, *bias; int ob, kind;
    if (o0 < 64)       { Wm = Wq; bias = bq; ob = o0;       kind = 0; }
    else if (o0 < 128) { Wm = Wk; bias = bk; ob = o0 - 64;  kind = 1; }
    else               { Wm = Wv; bias = bv; ob = o0 - 128; kind = 2; }

    v8f acc[6];
#pragma unroll
    for (int t = 0; t < 6; ++t) acc[t] = {};

    const float* wrow = Wm + (long)(ob + l16) * NC;
    for (int kk = 0; kk < NC; kk += 32) {
      v16bf af = load_afrag_w(wrow, kk, khf);   // A reused across 6 h-tiles
      // Batch all six B-fragment loads (12x ds_load_b128, one clause) so
      // LDS latency overlaps the wmma chain instead of stalling dscnt==0.
      v16bf bfr[6];
#pragma unroll
      for (int ct = 0; ct < 6; ++ct)
        bfr[ct] = *(const v16bf*)(XT + (ct * 16 + l16) * XP + kk + khf * 16);
#pragma unroll
      for (int ct = 0; ct < 6; ++ct)
        acc[ct] = __builtin_amdgcn_wmma_f32_16x16x32_bf16(
            false, af, false, bfr[ct], (short)0, acc[ct], false, false);
    }
    // C/D layout: element (M = r + 8*khf, N = l16)
#pragma unroll
    for (int r = 0; r < 8; ++r) {
      int m = r + 8 * khf;
      float bval = bias[ob + m];
#pragma unroll
      for (int ct = 0; ct < 6; ++ct) {
        int h = ct * 16 + l16;
        float v = acc[ct][r] + bval;
        if (kind == 0)      QT[h * QP + (ob + m)] = (__bf16)v;  // transposed
        else if (kind == 1) KT[h * QP + (ob + m)] = (__bf16)v;  // transposed
        else                VS[(ob + m) * VP + h] = (__bf16)v;  // row-major
      }
    }
  }
  __syncthreads();

  // ---------------- phase 2: E[i][j] = sum_c Q[i][c]*K[j][c] --------------
  for (int job = wave; job < 36; job += NWAVES) {
    int i0 = (job / 6) * 16, j0 = (job % 6) * 16;
    v16bf af[2], bf[2];
#pragma unroll
    for (int kt = 0; kt < 2; ++kt) {
      af[kt] = load_afrag_lds(QT + (i0 + l16) * QP, kt * 32, khf);
      bf[kt] = *(const v16bf*)(KT + (j0 + l16) * QP + kt * 32 + khf * 16);
    }
    v8f acc = {};
#pragma unroll
    for (int kt = 0; kt < 2; ++kt)
      acc = __builtin_amdgcn_wmma_f32_16x16x32_bf16(
          false, af[kt], false, bf[kt], (short)0, acc, false, false);
#pragma unroll
    for (int r = 0; r < 8; ++r)
      ES[(i0 + r + 8 * khf) * EP + j0 + l16] = acc[r];
  }
  __syncthreads();

  // ---------------- phase 3: row softmax over j (wave32 shuffles) ---------
  for (int i = wave; i < NH; i += NWAVES) {
    float v0 = ES[i * EP + lane];
    float v1 = ES[i * EP + lane + 32];
    float v2 = ES[i * EP + lane + 64];
    float m = fmaxf(v0, fmaxf(v1, v2));
#pragma unroll
    for (int off = 16; off > 0; off >>= 1) m = fmaxf(m, __shfl_xor(m, off, 32));
    float e0 = __expf(v0 - m), e1 = __expf(v1 - m), e2 = __expf(v2 - m);
    float s = e0 + e1 + e2;
#pragma unroll
    for (int off = 16; off > 0; off >>= 1) s += __shfl_xor(s, off, 32);
    float inv = 1.0f / s;
    AT[i * AP + lane]      = (__bf16)(e0 * inv);
    AT[i * AP + lane + 32] = (__bf16)(e1 * inv);
    AT[i * AP + lane + 64] = (__bf16)(e2 * inv);
  }
  __syncthreads();

  // ---------------- phase 4: Out[c][i] = sum_j V[c][j]*A[i][j] ------------
  for (int job = wave; job < 192; job += NWAVES) {
    int c0 = (job / 6) * 16;
    int i0 = (job % 6) * 16;
    v16bf af[3], bf[3];
#pragma unroll
    for (int kt = 0; kt < 3; ++kt) {
      af[kt] = load_afrag_lds(VS + (c0 + l16) * VP, kt * 32, khf);
      bf[kt] = *(const v16bf*)(AT + (i0 + l16) * AP + kt * 32 + khf * 16);
    }
    v8f acc = {};
#pragma unroll
    for (int kt = 0; kt < 3; ++kt)
      acc = __builtin_amdgcn_wmma_f32_16x16x32_bf16(
          false, af[kt], false, bf[kt], (short)0, acc, false, false);
#pragma unroll
    for (int r = 0; r < 8; ++r) {
      int c = c0 + r + 8 * khf;
      int h = i0 + l16;
      long gi = (((long)b * NC + c) * NH + h) * NW + w;
      out[gi] = g * acc[r] + x[gi];   // residual: second (and last) read of x
    }
  }
}

extern "C" void kernel_launch(void* const* d_in, const int* in_sizes, int n_in,
                              void* d_out, int out_size, void* d_ws, size_t ws_size,
                              hipStream_t stream) {
  (void)in_sizes; (void)n_in; (void)d_ws; (void)ws_size; (void)out_size;
  const float* x     = (const float*)d_in[0];
  const float* Wq    = (const float*)d_in[1];
  const float* bq    = (const float*)d_in[2];
  const float* Wk    = (const float*)d_in[3];
  const float* bk    = (const float*)d_in[4];
  const float* Wv    = (const float*)d_in[5];
  const float* bv    = (const float*)d_in[6];
  const float* gamma = (const float*)d_in[7];
  float* outp = (float*)d_out;

  // >64KB dynamic LDS opt-in (deterministic, capture-safe host call).
  hipFuncSetAttribute((const void*)cca_h_fused,
                      hipFuncAttributeMaxDynamicSharedMemorySize, SMEM_BYTES);

  dim3 grid(NW, NB);   // one workgroup per (w, b) column: 768 workgroups
  cca_h_fused<<<grid, NTHREADS, SMEM_BYTES, stream>>>(x, Wq, bq, Wk, bk, Wv, bv,
                                                      gamma, outp);
}
